// MultiHeadSelfAttention_22256520528322
// MI455X (gfx1250) — matrix operand order repaired and checked by reading, not compiled, not run on hardware
//
#include <hip/hip_runtime.h>
#include <hip/hip_bf16.h>
#include <math.h>

// ---------------------------------------------------------------------------
// MHA (causal), T=4096, D=1024, H=16, d_head=64.  bf16 WMMA, f32 accumulate.
// GEMMs: W tiles staged via the Tensor Data Mover (TENSORcnt), A tiles via
// global_load_async_to_lds (ASYNCcnt), both double-buffered against WMMA.
// Workspace layout (bytes from d_ws, 1MiB-aligned):
//   [ 0MB) xb bf16 4096x1024 | [ 8MB) Wq | [10MB) Wk | [12MB) Wv | [14MB) Wo
//   [16MB) Q | [24MB) K | [32MB) V | [40MB) Vt [H][64][T] | [48MB) attn  (56MB)
// ---------------------------------------------------------------------------

typedef __bf16 bf16;
typedef __attribute__((ext_vector_type(16))) __bf16 v16bf;
typedef __attribute__((ext_vector_type(8)))  __bf16 v8bf;
typedef __attribute__((ext_vector_type(8)))  float  v8f;
typedef __attribute__((ext_vector_type(4)))  unsigned u32x4;
typedef __attribute__((ext_vector_type(8)))  int      i32x8;
typedef __attribute__((ext_vector_type(4)))  int      i32x4;

#define DMODEL 1024
#define NHEADS 16
#define DHEAD  64
#define SEQT   4096

static __device__ __forceinline__ bf16 f2bf(float f) {
  union { float f; unsigned u; } v; v.f = f;
  unsigned r = v.u + 0x7FFFu + ((v.u >> 16) & 1u);   // round-to-nearest-even
  unsigned short h = (unsigned short)(r >> 16);
  bf16 b; __builtin_memcpy(&b, &h, 2); return b;
}

union AFrag { v16bf v; v8bf h[2]; };

// A fragment: 16x32 bf16 (MxK). lane<16 -> row M=lane, Ks {0..7,16..23};
// lane>=16 -> row M=lane-16, Ks {8..15,24..31}. src row-major, stride ld elems.
static __device__ __forceinline__ v16bf load_a(const bf16* __restrict__ src,
                                               int m0, int k0, int ld, int lane) {
  int hi = lane >> 4, mr = lane & 15;
  const bf16* p = src + (size_t)(m0 + mr) * ld + k0 + hi * 8;
  AFrag a;
  a.h[0] = *(const v8bf*)(p);
  a.h[1] = *(const v8bf*)(p + 16);
  return a.v;
}

// B fragment: 32x16 bf16 (KxN). Element (k,n) = src[(n0+n)*ld + k0 + k].
static __device__ __forceinline__ v16bf load_b(const bf16* __restrict__ src,
                                               int n0, int k0, int ld, int lane) {
  int hi = lane >> 4, nc = lane & 15;
  const bf16* p = src + (size_t)(n0 + nc) * ld + k0 + hi * 8;
  AFrag b;
  b.h[0] = *(const v8bf*)(p);
  b.h[1] = *(const v8bf*)(p + 16);
  return b.v;
}

static __device__ __forceinline__ v8f wmma_bf16(v16bf a, v16bf b, v8f c) {
  return __builtin_amdgcn_wmma_f32_16x16x32_bf16(false, a, false, b, (short)0, c,
                                                 false, false);
}

// CDNA5 async DMA: global -> LDS, 16 bytes, tracked by ASYNCcnt.
static __device__ __forceinline__ void async_b128(unsigned lds_off,
                                                  const void* gaddr) {
  asm volatile("global_load_async_to_lds_b128 %0, %1, off"
               :
               : "v"(lds_off), "v"(gaddr)
               : "memory");
}
static __device__ __forceinline__ void wait_asynccnt0() {
  asm volatile("s_wait_asynccnt 0x0" ::: "memory");
}
static __device__ __forceinline__ unsigned lds_off_of(const void* p) {
  return (unsigned)(uintptr_t)p;    // LDS aperture: low 32 bits = LDS byte addr
}

// CDNA5 Tensor Data Mover: one instruction DMAs a 2-D tile (rows x 32 bf16)
// from global into LDS.  D# per CDNA5 ISA.
// This toolchain exposes the 6-arg builtin (4 descriptor groups + spare + cpol).
static __device__ __forceinline__ void tdm_load_tile(unsigned lds_byte_off,
                                                     const void* gaddr,
                                                     int rows, int row_stride_elems) {
  unsigned long long ga = (unsigned long long)(uintptr_t)gaddr;
  u32x4 g0;
  g0[0] = 1u;                                        // count=1, no gather
  g0[1] = lds_byte_off;                              // lds_addr [63:32]
  g0[2] = (unsigned)ga;                              // global_addr [95:64]
  g0[3] = (unsigned)((ga >> 32) & 0x01FFFFFFu)       // global_addr [120:96]
          | 0x80000000u;                             // type=2 ("image") [127:126]
  i32x8 g1;
  g1[0] = 1 << 16;                                   // data_size=2B [17:16]
  g1[1] = 32 << 16;                                  // tensor_dim0=32 [79:48]
  g1[2] = rows << 16;                                // tensor_dim1 [111:80]
  g1[3] = 32 << 16;                                  // tile_dim0=32 [127:112]
  g1[4] = rows;                                      // tile_dim1 [143:128], tile_dim2=0
  g1[5] = row_stride_elems;                          // tensor_dim0_stride [207:160]
  g1[6] = 0;                                         // stride hi + dim1_stride lo
  g1[7] = 0;
  i32x4 g2 = {0, 0, 0, 0};                           // 2-D: dims 2..4 unused
  i32x4 g3 = {0, 0, 0, 0};
  i32x8 g4 = {0, 0, 0, 0, 0, 0, 0, 0};
  __builtin_amdgcn_tensor_load_to_lds(g0, g1, g2, g3, g4, 0);
}

// ---------------------------------------------------------------------------
__global__ void cvt_f32_bf16(const float* __restrict__ in, bf16* __restrict__ out,
                             int n) {
  int i = blockIdx.x * blockDim.x + threadIdx.x;
  if (i < n) out[i] = f2bf(in[i]);
}

// Vt[h][d][t] = V[t][h*64+d]
__global__ void transpose_v(const bf16* __restrict__ Vb, bf16* __restrict__ Vt) {
  int idx = blockIdx.x * blockDim.x + threadIdx.x;    // 0..T*1024
  int t = idx >> 10, c = idx & 1023;
  Vt[(size_t)c * SEQT + t] = Vb[idx];
}

// ---------------------------------------------------------------------------
// out[m][n] = sum_k A[m][k]*W[n][k] + bias[n].
// Workgroup = 4 waves = 128(M) x 64(N). Per 32-wide k-step:
//   * 64x32 W tile  -> LDS via ONE tdm instruction (wave 0), TENSORcnt
//   * 32x32 A tile  -> LDS per wave via async_to_lds_b128 (4 chunks/lane)
// Both double-buffered; compute reads only LDS (ds_load_b128) -> 8 WMMAs.
// ---------------------------------------------------------------------------
__global__ void gemm_bf16_wmma(const bf16* __restrict__ A, const bf16* __restrict__ W,
                               const float* __restrict__ bias, void* __restrict__ out,
                               int M, int N, int K, int out_f32) {
  __shared__ alignas(16) bf16 Wlds[2][64 * 32];       // 2 x 4KB (whole WG)
  __shared__ alignas(16) bf16 Alds[4][2][32 * 32];    // 2 x 2KB per wave
  const int lane = threadIdx.x;
  const int ty = threadIdx.y;
  const int n0 = blockIdx.x * 64;
  const int m0 = blockIdx.y * 128 + ty * 32;
  const int hi = lane >> 4, nc = lane & 15, mr = lane & 15;

  auto issue_w = [&](int bufi, int k0) {              // wave-uniform, ty==0 only
    tdm_load_tile(lds_off_of(&Wlds[bufi][0]),
                  W + (size_t)n0 * K + k0, 64, K);
  };
  auto issue_a = [&](int bufi, int k0) {              // 4 x 16B per lane
#pragma unroll
    for (int i = 0; i < 4; ++i) {
      int c = lane + 32 * i;                          // 0..127 chunk id
      int row = c >> 2, off = (c & 3) * 8;
      async_b128(lds_off_of(&Alds[ty][bufi][row * 32 + off]),
                 A + (size_t)(m0 + row) * K + k0 + off);
    }
  };

  v8f acc[2][4];
#pragma unroll
  for (int s = 0; s < 2; ++s)
#pragma unroll
    for (int j = 0; j < 4; ++j) acc[s][j] = v8f{};

  if (ty == 0) issue_w(0, 0);
  issue_a(0, 0);
  if (ty == 0) __builtin_amdgcn_s_wait_tensorcnt(0);
  wait_asynccnt0();
  __syncthreads();

  int buf = 0;
  for (int k0 = 0; k0 < K; k0 += 32, buf ^= 1) {
    const bool more = (k0 + 32) < K;
    if (more) {                                       // next tiles in flight
      if (ty == 0) issue_w(buf ^ 1, k0 + 32);
      issue_a(buf ^ 1, k0 + 32);
    }

    // A fragments for the wave's two 16-row subtiles, from LDS
    AFrag a0, a1;
    {
      const bf16* p0 = &Alds[ty][buf][mr * 32 + hi * 8];
      const bf16* p1 = &Alds[ty][buf][(16 + mr) * 32 + hi * 8];
      a0.h[0] = *(const v8bf*)(p0); a0.h[1] = *(const v8bf*)(p0 + 16);
      a1.h[0] = *(const v8bf*)(p1); a1.h[1] = *(const v8bf*)(p1 + 16);
    }
#pragma unroll
    for (int j = 0; j < 4; ++j) {
      const bf16* p = &Wlds[buf][(j * 16 + nc) * 32 + hi * 8];
      AFrag b;
      b.h[0] = *(const v8bf*)(p);
      b.h[1] = *(const v8bf*)(p + 16);
      acc[0][j] = wmma_bf16(a0.v, b.v, acc[0][j]);
      acc[1][j] = wmma_bf16(a1.v, b.v, acc[1][j]);
    }
    if (more) {
      if (ty == 0) __builtin_amdgcn_s_wait_tensorcnt(0);
      wait_asynccnt0();
      __syncthreads();
    }
  }

#pragma unroll
  for (int s = 0; s < 2; ++s)
#pragma unroll
    for (int j = 0; j < 4; ++j) {
      int n = n0 + j * 16 + nc;
      float bval = bias[n];
#pragma unroll
      for (int r = 0; r < 8; ++r) {
        int m = m0 + s * 16 + r + 8 * hi;             // C/D layout row
        float v = acc[s][j][r] + bval;
        if (out_f32) ((float*)out)[(size_t)m * N + n] = v;
        else         ((bf16*)out)[(size_t)m * N + n] = f2bf(v);
      }
    }
}

// ---------------------------------------------------------------------------
// Flash-attention: one wave per (head, 16-row query tile). Streams keys 32 at
// a time; scores via 2x2 WMMA, online softmax with 16-lane shfl reductions, P
// round-trips LDS to re-fragment as a WMMA A operand, then 4 WMMAs for P.V.
// ---------------------------------------------------------------------------
__global__ void attn_wmma(const bf16* __restrict__ Q, const bf16* __restrict__ Kb,
                          const bf16* __restrict__ Vt, bf16* __restrict__ O) {
  __shared__ alignas(16) bf16 Plds[4][16 * 32];
  const int lane = threadIdx.x;
  const int ty = threadIdx.y;
  const int tile = blockIdx.x * blockDim.y + ty;
  const int h = tile >> 8;
  const int q0 = (tile & 255) * 16;
  const int hi = lane >> 4, nc = lane & 15;
  const float scale = 0.125f;                 // 1/sqrt(64)
  const float NEG = -3.0e38f;

  v16bf aq0 = load_a(Q, q0, h * DHEAD, DMODEL, lane);
  v16bf aq1 = load_a(Q, q0, h * DHEAD + 32, DMODEL, lane);

  v8f o[4] = {v8f{}, v8f{}, v8f{}, v8f{}};
  float mi[8], li[8];
#pragma unroll
  for (int r = 0; r < 8; ++r) { mi[r] = NEG; li[r] = 0.0f; }

  const bf16* Vth = Vt + (size_t)h * DHEAD * SEQT;
  const int kend = q0 + 16;                   // causal: keys < kend

  for (int kb = 0; kb < kend; kb += 32) {
    v8f s0 = {}, s1 = {};
    {
      v16bf b = load_b(Kb, kb, h * DHEAD, DMODEL, lane);
      s0 = wmma_bf16(aq0, b, s0);
      b = load_b(Kb, kb, h * DHEAD + 32, DMODEL, lane);
      s0 = wmma_bf16(aq1, b, s0);
    }
    if (kb + 16 < kend) {
      v16bf b = load_b(Kb, kb + 16, h * DHEAD, DMODEL, lane);
      s1 = wmma_bf16(aq0, b, s1);
      b = load_b(Kb, kb + 16, h * DHEAD + 32, DMODEL, lane);
      s1 = wmma_bf16(aq1, b, s1);
    }

#pragma unroll
    for (int r = 0; r < 8; ++r) {
      int qg = q0 + r + 8 * hi;
      float v0 = s0[r] * scale;
      float v1 = s1[r] * scale;
      if (kb + nc > qg)      v0 = NEG;
      if (kb + 16 + nc > qg) v1 = NEG;

      float mx = fmaxf(v0, v1);
#pragma unroll
      for (int off = 1; off < 16; off <<= 1)
        mx = fmaxf(mx, __shfl_xor(mx, off, 32));   // reduce within 16-lane half
      float newm = fmaxf(mi[r], mx);
      float alpha = __expf(mi[r] - newm);
      float p0 = __expf(v0 - newm);
      float p1 = __expf(v1 - newm);
      float sum = p0 + p1;
#pragma unroll
      for (int off = 1; off < 16; off <<= 1)
        sum += __shfl_xor(sum, off, 32);
      li[r] = li[r] * alpha + sum;
      mi[r] = newm;
#pragma unroll
      for (int j = 0; j < 4; ++j) o[j][r] *= alpha;

      int row = r + 8 * hi;
      Plds[ty][row * 32 + nc]      = f2bf(p0);
      Plds[ty][row * 32 + 16 + nc] = f2bf(p1);
    }

    AFrag ap;
    const bf16* pp = &Plds[ty][(lane & 15) * 32 + hi * 8];
    ap.h[0] = *(const v8bf*)(pp);
    ap.h[1] = *(const v8bf*)(pp + 16);

#pragma unroll
    for (int j = 0; j < 4; ++j) {
      v16bf bv = load_b(Vth, j * 16, kb, SEQT, lane);  // (k=token, n=d)
      o[j] = wmma_bf16(ap.v, bv, o[j]);
    }
  }

#pragma unroll
  for (int j = 0; j < 4; ++j) {
#pragma unroll
    for (int r = 0; r < 8; ++r) {
      int m = q0 + r + 8 * hi;
      int n = h * DHEAD + j * 16 + nc;
      O[(size_t)m * DMODEL + n] = f2bf(o[j][r] / li[r]);
    }
  }
}

// ---------------------------------------------------------------------------
extern "C" void kernel_launch(void* const* d_in, const int* in_sizes, int n_in,
                              void* d_out, int out_size, void* d_ws, size_t ws_size,
                              hipStream_t stream) {
  const float* x  = (const float*)d_in[0];
  const float* Wq = (const float*)d_in[1];
  const float* bq = (const float*)d_in[2];
  const float* Wk = (const float*)d_in[3];
  const float* bk = (const float*)d_in[4];
  const float* Wv = (const float*)d_in[5];
  const float* bv = (const float*)d_in[6];
  const float* Wo = (const float*)d_in[7];
  const float* bo = (const float*)d_in[8];
  // d_in[9] = mask (causal mask computed analytically in-kernel)

  char* w = (char*)d_ws;
  const size_t MB = 1u << 20;
  bf16* xb  = (bf16*)(w + 0 * MB);
  bf16* wqb = (bf16*)(w + 8 * MB);
  bf16* wkb = (bf16*)(w + 10 * MB);
  bf16* wvb = (bf16*)(w + 12 * MB);
  bf16* wob = (bf16*)(w + 14 * MB);
  bf16* Qb  = (bf16*)(w + 16 * MB);
  bf16* Kb  = (bf16*)(w + 24 * MB);
  bf16* Vb  = (bf16*)(w + 32 * MB);
  bf16* Vt  = (bf16*)(w + 40 * MB);
  bf16* Ab  = (bf16*)(w + 48 * MB);

  const int NX = SEQT * DMODEL;       // 4096*1024
  const int NW = DMODEL * DMODEL;     // 1024*1024
  cvt_f32_bf16<<<(NX + 255) / 256, 256, 0, stream>>>(x,  xb,  NX);
  cvt_f32_bf16<<<(NW + 255) / 256, 256, 0, stream>>>(Wq, wqb, NW);
  cvt_f32_bf16<<<(NW + 255) / 256, 256, 0, stream>>>(Wk, wkb, NW);
  cvt_f32_bf16<<<(NW + 255) / 256, 256, 0, stream>>>(Wv, wvb, NW);
  cvt_f32_bf16<<<(NW + 255) / 256, 256, 0, stream>>>(Wo, wob, NW);

  dim3 gblk(32, 4);                                  // 4 waves
  dim3 ggrd(DMODEL / 64, SEQT / 128);                // (16, 32)
  gemm_bf16_wmma<<<ggrd, gblk, 0, stream>>>(xb, wqb, bq, Qb, SEQT, DMODEL, DMODEL, 0);
  gemm_bf16_wmma<<<ggrd, gblk, 0, stream>>>(xb, wkb, bk, Kb, SEQT, DMODEL, DMODEL, 0);
  gemm_bf16_wmma<<<ggrd, gblk, 0, stream>>>(xb, wvb, bv, Vb, SEQT, DMODEL, DMODEL, 0);

  transpose_v<<<NX / 256, 256, 0, stream>>>(Vb, Vt);

  attn_wmma<<<dim3((NHEADS * 256) / 4), dim3(32, 4), 0, stream>>>(Qb, Kb, Vt, Ab);

  gemm_bf16_wmma<<<ggrd, gblk, 0, stream>>>(Ab, wob, bo, d_out, SEQT, DMODEL, DMODEL, 1);
}
